// LearnableRetriever_84670985274058
// MI455X (gfx1250) — matrix operand order repaired and checked by reading, not compiled, not run on hardware
//
#include <hip/hip_runtime.h>
#include <hip/hip_bf16.h>
#include <math.h>

// ---------------------------------------------------------------------------
// Types for CDNA5 WMMA (wave32, V_WMMA_F32_16X16X32_BF16)
// ---------------------------------------------------------------------------
typedef __bf16 v16bf __attribute__((ext_vector_type(16)));
typedef float  v8f   __attribute__((ext_vector_type(8)));
typedef unsigned int v4u __attribute__((ext_vector_type(4)));
// GCC-style vector for the async-LDS builtin prototype (clang "V4i").
typedef int v4i_vs __attribute__((vector_size(16)));

// ---------------------------------------------------------------------------
// CDNA5 async global->LDS copy (ASYNCcnt-tracked, no VGPR data path).
// Signature (from hipcc diagnostic): param0 = addrspace(1) v4i*, 4 args.
// Guarded: falls back to a synchronous v4u copy if the builtin is absent.
// ---------------------------------------------------------------------------
#if defined(__has_builtin)
#if __has_builtin(__builtin_amdgcn_global_load_async_to_lds_b128)
#define HAVE_ASYNC_LDS 1
#endif
#endif
#ifndef HAVE_ASYNC_LDS
#define HAVE_ASYNC_LDS 0
#endif

#if HAVE_ASYNC_LDS
__device__ __forceinline__ void async_ld16(void* lds_dst, const void* g_src) {
    __builtin_amdgcn_global_load_async_to_lds_b128(
        (__attribute__((address_space(1))) v4i_vs*)g_src,
        (__attribute__((address_space(3))) v4i_vs*)lds_dst,
        0, 0);
}
#if __has_builtin(__builtin_amdgcn_s_wait_asynccnt)
#define WAIT_ASYNCCNT(n) __builtin_amdgcn_s_wait_asynccnt(n)
#else
#define WAIT_ASYNCCNT(n) asm volatile("s_wait_asynccnt %0" :: "i"(n) : "memory")
#endif
#else
#define WAIT_ASYNCCNT(n) ((void)0)
#endif

// float -> bf16, round-to-nearest-even (bit-level; inputs are normal floats)
__device__ __forceinline__ unsigned short f2bf(float x) {
    unsigned int u = __float_as_uint(x);
    u += 0x7FFFu + ((u >> 16) & 1u);
    return (unsigned short)(u >> 16);
}

// Load a 16-element bf16 fragment as two 16-byte chunks (ds_load_b128 /
// global_load_b128 depending on the pointer's address space).
__device__ __forceinline__ v16bf ld_frag2(const unsigned short* p0,
                                          const unsigned short* p1) {
    union { v16bf v; v4u q[2]; } u;
    u.q[0] = *reinterpret_cast<const v4u*>(p0);
    u.q[1] = *reinterpret_cast<const v4u*>(p1);
    return u.v;
}

__device__ __forceinline__ void top3_insert(float x, int c,
                                            float& v0, int& i0,
                                            float& v1, int& i1,
                                            float& v2, int& i2) {
    if (x > v2) {
        if (x > v1) {
            if (x > v0) { v2 = v1; i2 = i1; v1 = v0; i1 = i0; v0 = x; i0 = c; }
            else        { v2 = v1; i2 = i1; v1 = x;  i1 = c; }
        } else          { v2 = x;  i2 = c; }
    }
}

// ---------------------------------------------------------------------------
// Kernel 1/2: out_bf16[M,N] = act(A[M,K] @ B[K,N] + bias[N])
//   A is f32 (ABF16=false) or bf16 (ABF16=true), B/bias f32, out bf16.
//   64x64 output tile per block, K streamed through LDS in 128-chunks.
// ---------------------------------------------------------------------------
#define LDSA_STRIDE 136   // 128 + 8 u16 pad (272B = 17*16B, keeps 16B align)

template<bool ABF16, bool RELU>
__global__ __launch_bounds__(256)
void wmma_gemm_bias(const float* __restrict__ A32,
                    const unsigned short* __restrict__ A16,
                    const float* __restrict__ B32,
                    const float* __restrict__ bias,
                    unsigned short* __restrict__ outBf,
                    int M, int N, int K) {
    __shared__ unsigned short ldsA[64 * LDSA_STRIDE];
    __shared__ unsigned short ldsBT[64 * LDSA_STRIDE];

    const int tid = threadIdx.x;
    const int w   = tid >> 5;
    const int l   = tid & 31;
    const int lh  = l >> 4;      // lane half (K striping)
    const int ll  = l & 15;      // lane low (M for A, N for B)
    const int mt  = w & 3;
    const int ntb = (w >> 2) * 2;
    const int mBase = blockIdx.x * 64;
    const int nBase = blockIdx.y * 64;

    v8f acc0 = {}, acc1 = {};

    const int nChunks = K >> 7;  // K / 128
    for (int kb = 0; kb < nChunks; ++kb) {
        const int kOff = kb << 7;
        // Stage A chunk: ldsA[r][k] = bf16(A[mBase+r][kOff+k]), coalesced on k.
        for (int i = tid; i < 64 * 128; i += 256) {
            int r = i >> 7, k = i & 127;
            unsigned short v;
            if (ABF16) v = A16[(size_t)(mBase + r) * K + kOff + k];
            else       v = f2bf(A32[(size_t)(mBase + r) * K + kOff + k]);
            ldsA[r * LDSA_STRIDE + k] = v;
        }
        // Stage B transposed: ldsBT[n][k] = bf16(B[kOff+k][nBase+n]),
        // coalesced on n (64 consecutive floats per step).
        for (int i = tid; i < 64 * 128; i += 256) {
            int j = i & 63, k = i >> 6;
            ldsBT[j * LDSA_STRIDE + k] =
                f2bf(B32[(size_t)(kOff + k) * N + nBase + j]);
        }
        __syncthreads();
        #pragma unroll
        for (int kc = 0; kc < 4; ++kc) {
            // A fragment: lane ll = row, K contiguous (lo half: 0..15,
            // hi half: 16..31), matching B.
            const unsigned short* ap = &ldsA[(mt * 16 + ll) * LDSA_STRIDE + kc * 32];
            v16bf a = ld_frag2(ap + lh * 16, ap + lh * 16 + 8);
            // B fragments: lane ll = column, K contiguous (lo: 0..15, hi: 16..31).
            const unsigned short* bp0 = &ldsBT[((ntb + 0) * 16 + ll) * LDSA_STRIDE + kc * 32 + lh * 16];
            const unsigned short* bp1 = &ldsBT[((ntb + 1) * 16 + ll) * LDSA_STRIDE + kc * 32 + lh * 16];
            v16bf b0 = ld_frag2(bp0, bp0 + 8);
            v16bf b1 = ld_frag2(bp1, bp1 + 8);
            acc0 = __builtin_amdgcn_wmma_f32_16x16x32_bf16(false, a, false, b0,
                                                           (short)0, acc0, false, false);
            acc1 = __builtin_amdgcn_wmma_f32_16x16x32_bf16(false, a, false, b1,
                                                           (short)0, acc1, false, false);
        }
        __syncthreads();
    }

    // Epilogue: bias (+ReLU) + f32->bf16 store.
    #pragma unroll
    for (int j = 0; j < 2; ++j) {
        v8f acc = j ? acc1 : acc0;
        const int gCol = nBase + (ntb + j) * 16 + ll;
        const float bv = bias[gCol];
        #pragma unroll
        for (int r = 0; r < 8; ++r) {
            const int gRow = mBase + mt * 16 + r + lh * 8;  // C/D layout: VGPR r -> M=r / r+8
            float v = acc[r] + bv;
            if (RELU) v = fmaxf(v, 0.0f);
            outBf[(size_t)gRow * N + gCol] = f2bf(v);
        }
    }
}

// ---------------------------------------------------------------------------
// Kernel 3: fused  sim = P @ P^T  ->  per-row top-3  ->  softmax  ->
//           gather sess_emb + weighted sum. P = sess_proj in bf16 (8192x512).
// Block: 64 rows, 256 threads (8 waves). A fragments held in registers for
// the full K=512 (16 x v16bf). 256 iterations over 32-column tiles of B,
// double-buffered via GLOBAL_LOAD_ASYNC_TO_LDS_B128 (ASYNCcnt), two
// independent WMMA accumulator chains per wave.
// ---------------------------------------------------------------------------
#define B_ROWS     32
#define B_STRIDE   520   // 512 + 8 u16 pad (1040B = 65*16B)
#define SIM_STRIDE 33
#define STAGE_LOADS_PER_THREAD 8   // 32 rows * 64 chunks / 256 threads

__device__ __forceinline__ void issue_stage(unsigned short* __restrict__ buf,
                                            const unsigned short* __restrict__ pbf,
                                            int colBase, int tid) {
#if HAVE_ASYNC_LDS
    #pragma unroll
    for (int s = 0; s < STAGE_LOADS_PER_THREAD; ++s) {
        const int i = tid + s * 256;
        const int j = i >> 6, c = i & 63;
        async_ld16(&buf[j * B_STRIDE + c * 8],
                   pbf + ((size_t)(colBase + j) * 512 + (size_t)c * 8));
    }
#else
    const v4u* src = reinterpret_cast<const v4u*>(pbf + (size_t)colBase * 512);
    #pragma unroll
    for (int s = 0; s < STAGE_LOADS_PER_THREAD; ++s) {
        const int i = tid + s * 256;
        const int j = i >> 6, c = i & 63;
        *reinterpret_cast<v4u*>(&buf[j * B_STRIDE + c * 8]) = src[(size_t)j * 64 + c];
    }
#endif
}

__global__ __launch_bounds__(256)
void sim_topk_kernel(const unsigned short* __restrict__ pbf,   // 8192x512 bf16
                     const float* __restrict__ sess_emb,       // 8192x512 f32
                     float* __restrict__ out_topk,             // 8192x3x512
                     float* __restrict__ out_nb,               // 8192x512
                     float* __restrict__ out_cos) {            // 8192x3
    __shared__ unsigned short ldsB[2][B_ROWS * B_STRIDE];
    __shared__ float simT[64 * SIM_STRIDE];
    __shared__ float candV[64 * 2 * 3];
    __shared__ int   candI[64 * 2 * 3];
    __shared__ float rowW[64 * 3];
    __shared__ int   rowI[64 * 3];

    const int tid = threadIdx.x;
    const int w   = tid >> 5;
    const int l   = tid & 31;
    const int lh  = l >> 4;
    const int ll  = l & 15;
    const int mt  = w & 3;       // 4 row tiles
    const int nt  = w >> 2;      // 2 col tiles -> 8 waves, 1 tile each
    const int rowBase = blockIdx.x * 64;

    // Preload A fragments for all 16 K-steps straight from global (L2-resident).
    v16bf afrag[16];
    {
        const unsigned short* arow = pbf + (size_t)(rowBase + mt * 16 + ll) * 512;
        #pragma unroll
        for (int kc = 0; kc < 16; ++kc)
            afrag[kc] = ld_frag2(arow + kc * 32 + lh * 16,
                                 arow + kc * 32 + lh * 16 + 8);
    }

    // Private running top-3: thread t (<128) owns (row = t&63, stripe = t>>6).
    float pv0 = -INFINITY, pv1 = -INFINITY, pv2 = -INFINITY;
    int   pi0 = 0, pi1 = 0, pi2 = 0;
    const int row_s  = tid & 63;
    const int half_s = tid >> 6;

    // Prime the pipeline with tile 0.
    issue_stage(ldsB[0], pbf, 0, tid);

    for (int ct = 0; ct < 256; ++ct) {
        const int cur = ct & 1;
        const int colBase = ct * B_ROWS;

        // Issue the NEXT tile into the alternate buffer, then retire the
        // current tile's async batch (in-order per thread: wait <= 8 leaves
        // exactly the newer batch outstanding).
        if (ct + 1 < 256) {
            issue_stage(ldsB[cur ^ 1], pbf, (ct + 1) * B_ROWS, tid);
            WAIT_ASYNCCNT(STAGE_LOADS_PER_THREAD);
        } else {
            WAIT_ASYNCCNT(0);
        }
        __syncthreads();

        // 64x32 sim tile; two independent accumulator chains (even/odd kc).
        const unsigned short* bbase = &ldsB[cur][(nt * 16 + ll) * B_STRIDE];
        v8f accA = {}, accB = {};
        #pragma unroll
        for (int kc = 0; kc < 16; kc += 2) {
            const unsigned short* bp0 = bbase + kc * 32 + lh * 16;
            const unsigned short* bp1 = bbase + (kc + 1) * 32 + lh * 16;
            v16bf b0 = ld_frag2(bp0, bp0 + 8);
            v16bf b1 = ld_frag2(bp1, bp1 + 8);
            accA = __builtin_amdgcn_wmma_f32_16x16x32_bf16(false, afrag[kc], false, b0,
                                                           (short)0, accA, false, false);
            accB = __builtin_amdgcn_wmma_f32_16x16x32_bf16(false, afrag[kc + 1], false, b1,
                                                           (short)0, accB, false, false);
        }
        #pragma unroll
        for (int r = 0; r < 8; ++r)
            simT[(mt * 16 + r + lh * 8) * SIM_STRIDE + nt * 16 + ll] = accA[r] + accB[r];
        __syncthreads();

        // 128 threads scan: each owns one row x one 16-col stripe.
        if (tid < 128) {
            const float* sp = &simT[row_s * SIM_STRIDE + half_s * 16];
            const int cg = colBase + half_s * 16;
            #pragma unroll 4
            for (int i = 0; i < 16; ++i)
                top3_insert(sp[i], cg + i, pv0, pi0, pv1, pi1, pv2, pi2);
        }
        __syncthreads();
    }

    // Merge the two stripes per row, softmax over top-3.
    if (tid < 128) {
        const int b = (row_s * 2 + half_s) * 3;
        candV[b] = pv0; candV[b + 1] = pv1; candV[b + 2] = pv2;
        candI[b] = pi0; candI[b + 1] = pi1; candI[b + 2] = pi2;
    }
    __syncthreads();
    if (tid < 64) {
        float v0 = -INFINITY, v1 = -INFINITY, v2 = -INFINITY;
        int   i0 = 0, i1 = 0, i2 = 0;
        #pragma unroll
        for (int h = 0; h < 2; ++h)
            #pragma unroll
            for (int j = 0; j < 3; ++j) {
                const int b = (tid * 2 + h) * 3 + j;
                top3_insert(candV[b], candI[b], v0, i0, v1, i1, v2, i2);
            }
        const float e1 = __expf(v1 - v0);
        const float e2 = __expf(v2 - v0);
        const float inv = 1.0f / (1.0f + e1 + e2);
        const float w0 = inv, w1 = e1 * inv, w2 = e2 * inv;
        rowW[tid * 3 + 0] = w0; rowW[tid * 3 + 1] = w1; rowW[tid * 3 + 2] = w2;
        rowI[tid * 3 + 0] = i0; rowI[tid * 3 + 1] = i1; rowI[tid * 3 + 2] = i2;
        const size_t gRow = rowBase + tid;
        out_cos[gRow * 3 + 0] = w0;
        out_cos[gRow * 3 + 1] = w1;
        out_cos[gRow * 3 + 2] = w2;
    }
    __syncthreads();

    // Gather (exact f32 sess_emb rows) + weighted sum, float4, 4 threads/row.
    {
        const int row = tid >> 2;
        const int q   = tid & 3;
        const size_t gRow = rowBase + row;
        const float w0 = rowW[row * 3 + 0];
        const float w1 = rowW[row * 3 + 1];
        const float w2 = rowW[row * 3 + 2];
        const float4* s0 = reinterpret_cast<const float4*>(sess_emb + (size_t)rowI[row * 3 + 0] * 512);
        const float4* s1 = reinterpret_cast<const float4*>(sess_emb + (size_t)rowI[row * 3 + 1] * 512);
        const float4* s2 = reinterpret_cast<const float4*>(sess_emb + (size_t)rowI[row * 3 + 2] * 512);
        float4* t0 = reinterpret_cast<float4*>(out_topk + (gRow * 3 + 0) * 512);
        float4* t1 = reinterpret_cast<float4*>(out_topk + (gRow * 3 + 1) * 512);
        float4* t2 = reinterpret_cast<float4*>(out_topk + (gRow * 3 + 2) * 512);
        float4* nb = reinterpret_cast<float4*>(out_nb + gRow * 512);
        for (int c = q * 32; c < q * 32 + 32; ++c) {
            const float4 a = s0[c];
            const float4 b = s1[c];
            const float4 d = s2[c];
            t0[c] = a; t1[c] = b; t2[c] = d;
            float4 n;
            n.x = w0 * a.x + w1 * b.x + w2 * d.x;
            n.y = w0 * a.y + w1 * b.y + w2 * d.y;
            n.z = w0 * a.z + w1 * b.z + w2 * d.z;
            n.w = w0 * a.w + w1 * b.w + w2 * d.w;
            nb[c] = n;
        }
    }
}

// ---------------------------------------------------------------------------
// Launch: GEMM1 (ReLU) -> GEMM2 -> fused sim/top-k/softmax/gather.
// Workspace: hbf (8192x256 bf16, 4 MB) | pbf (8192x512 bf16, 8 MB) = 12 MB.
// ---------------------------------------------------------------------------
extern "C" void kernel_launch(void* const* d_in, const int* in_sizes, int n_in,
                              void* d_out, int out_size, void* d_ws, size_t ws_size,
                              hipStream_t stream) {
    const float* sess = (const float*)d_in[0];
    // d_in[1] = pool_emb: unused by the reference computation.
    const float* W1 = (const float*)d_in[2];
    const float* b1 = (const float*)d_in[3];
    const float* W2 = (const float*)d_in[4];
    const float* b2 = (const float*)d_in[5];

    const int B = 8192, D = 512, H = 256;

    unsigned short* hbf = (unsigned short*)d_ws;               // B x H bf16
    unsigned short* pbf = hbf + (size_t)B * H;                 // B x D bf16

    float* out_topk = (float*)d_out;                           // B x 3 x D
    float* out_nb   = out_topk + (size_t)B * 3 * D;            // B x D
    float* out_cos  = out_nb + (size_t)B * D;                  // B x 3

    // h = relu(sess_emb @ W1 + b1)
    wmma_gemm_bias<false, true><<<dim3(B / 64, H / 64), 256, 0, stream>>>(
        sess, nullptr, W1, b1, hbf, B, H, D);
    // sess_proj = h @ W2 + b2
    wmma_gemm_bias<true, false><<<dim3(B / 64, D / 64), 256, 0, stream>>>(
        nullptr, hbf, W2, b2, pbf, B, D, H);
    // sim -> top3 -> softmax -> gather/weighted sum
    sim_topk_kernel<<<B / 64, 256, 0, stream>>>(
        pbf, sess, out_topk, out_nb, out_cos);
}
